// GNNCritic6_27195732918304
// MI455X (gfx1250) — matrix operand order repaired and checked by reading, not compiled, not run on hardware
//
#include <hip/hip_runtime.h>
#include <hip/hip_bf16.h>
#include <stdint.h>

typedef __attribute__((ext_vector_type(16))) __bf16 v16bf;
typedef __attribute__((ext_vector_type(8)))  float  v8f;
typedef __attribute__((ext_vector_type(2)))  float  v2f;
typedef __attribute__((ext_vector_type(4)))  int    v4i;

#define N_NODES 204800
#define C_IN    64
#define E_EDGES 3276800
#define B_GRAPH 2048
#define R_REG   100
#define H_DIM   256
#define K1PAD   192   // 64 (features) + 100 (action) padded to 6*32

// ---- gfx1250 async global->LDS path (ASYNCcnt), guarded so compile never breaks
#if defined(__has_builtin)
#if __has_builtin(__builtin_amdgcn_global_load_async_to_lds_b128) && __has_builtin(__builtin_amdgcn_s_wait_asynccnt)
#define USE_ASYNC_LDS 1
#endif
#endif
#ifndef USE_ASYNC_LDS
#define USE_ASYNC_LDS 0
#endif

typedef __attribute__((address_space(1))) v4i gv4i;
typedef __attribute__((address_space(3))) v4i lv4i;

__device__ __forceinline__ void async_copy16(const void* g, void* l) {
#if USE_ASYNC_LDS
  __builtin_amdgcn_global_load_async_to_lds_b128((gv4i*)g, (lv4i*)l, 0, 0);
#else
  *(uint4*)l = *(const uint4*)g;
#endif
}

__device__ __forceinline__ void async_wait() {
#if USE_ASYNC_LDS
  __builtin_amdgcn_s_wait_asynccnt(0);
#endif
}

__device__ __forceinline__ unsigned short f2bf(float f) {
  unsigned u = __float_as_uint(f);
  u += 0x7FFFu + ((u >> 16) & 1u);   // round-to-nearest-even
  return (unsigned short)(u >> 16);
}

union ABfrag { v16bf v; unsigned u[8]; };

// ---------------------------------------------------------------- init
__global__ void k_init(float* __restrict__ agg, float* __restrict__ deg) {
  int i = blockIdx.x * 256 + threadIdx.x;
  if (i < N_NODES * C_IN) agg[i] = 0.f;
  if (i < N_NODES)        deg[i] = 1.0f;   // self-loop
}

// ---------------------------------------------------------------- degree
__global__ void k_deg(const long long* __restrict__ dst, float* __restrict__ deg) {
  int e = blockIdx.x * 256 + threadIdx.x;
  if (e < E_EDGES) atomicAdd(&deg[(int)dst[e]], 1.0f);
}

__global__ void k_dinv(float* __restrict__ deg) {
  int n = blockIdx.x * 256 + threadIdx.x;
  if (n < N_NODES) deg[n] = rsqrtf(deg[n]);
}

// ---------------------------------------------------------------- xw = state @ W_gcn  (fp32 WMMA 16x16x4)
__global__ __launch_bounds__(256) void k_xw(const float* __restrict__ state,
                                            const float* __restrict__ Wg,
                                            float* __restrict__ xw) {
  __shared__ float As[32][68];   // 272B rows: 16B-aligned, conflict-free (68 % 64 = 4)
  __shared__ float Bs[64][64];   // W_gcn
  int tid = threadIdx.x, lane = tid & 31, wave = tid >> 5;
  int rowBase = blockIdx.x * 32;

  for (int i = tid; i < 32 * 16; i += 256) {          // 16B chunks: 32 rows x 16
    int r = i >> 4, q = i & 15;
    async_copy16(state + (size_t)(rowBase + r) * C_IN + q * 4, &As[r][q * 4]);
  }
  for (int i = tid; i < 64 * 16; i += 256) {          // W_gcn straight copy
    async_copy16(Wg + (size_t)i * 4, &Bs[0][0] + (size_t)i * 4);
  }
  async_wait();
  __syncthreads();

  int mt = wave >> 2, nt = wave & 3;          // 2 x 4 tile grid over 32x64
  int lm = mt * 16 + (lane & 15);
  int n  = nt * 16 + (lane & 15);
  int kh = (lane < 16) ? 0 : 2;               // A/B K-half per ISA layout
  v8f acc = {0, 0, 0, 0, 0, 0, 0, 0};
  #pragma unroll
  for (int ks = 0; ks < 16; ++ks) {
    int k0 = ks * 4 + kh;
    v2f a = {As[lm][k0], As[lm][k0 + 1]};
    v2f b = {Bs[k0][n],  Bs[k0 + 1][n]};
    acc = __builtin_amdgcn_wmma_f32_16x16x4_f32(false, a, false, b, (short)0, acc, false, false);
  }
  #pragma unroll
  for (int r = 0; r < 8; ++r) {
    int gm = rowBase + mt * 16 + r + ((lane < 16) ? 0 : 8);
    xw[(size_t)gm * C_IN + n] = acc[r];
  }
}

// ---------------------------------------------------------------- edge scatter (4 threads / edge)
__global__ void k_scatter(const long long* __restrict__ src, const long long* __restrict__ dst,
                          const float* __restrict__ dinv, const float* __restrict__ xw,
                          float* __restrict__ agg) {
  int idx = blockIdx.x * 256 + threadIdx.x;
  int e = idx >> 2, q = idx & 3;
  if (e >= E_EDGES) return;
  int s = (int)src[e], d = (int)dst[e];
  float w = dinv[s] * dinv[d];
  const float4* xs = (const float4*)(xw + (size_t)s * C_IN + q * 16);
  float* ag = agg + (size_t)d * C_IN + q * 16;
  #pragma unroll
  for (int v = 0; v < 4; ++v) {
    float4 x = xs[v];
    atomicAdd(ag + v * 4 + 0, x.x * w);
    atomicAdd(ag + v * 4 + 1, x.y * w);
    atomicAdd(ag + v * 4 + 2, x.z * w);
    atomicAdd(ag + v * 4 + 3, x.w * w);
  }
}

// ---------------------------------------------------------------- build bf16 [relu(gcn)+state | action | 0] (K padded to 192)
__global__ void k_build_acat(const float* __restrict__ agg, const float* __restrict__ xw,
                             const float* __restrict__ dinv, const float* __restrict__ b_gcn,
                             const float* __restrict__ state, const float* __restrict__ action,
                             unsigned short* __restrict__ Acat) {
  int i = blockIdx.x * 256 + threadIdx.x;
  if (i >= N_NODES * K1PAD) return;
  int n = i / K1PAD, c = i % K1PAD;
  float val;
  if (c < C_IN) {
    float di = dinv[n];
    // self-loop message folded in: xw[n]*dinv[n]^2
    float g = agg[(size_t)n * C_IN + c] + xw[(size_t)n * C_IN + c] * di * di + b_gcn[c];
    val = fmaxf(g, 0.f) + state[(size_t)n * C_IN + c];
  } else if (c < C_IN + R_REG) {
    val = action[(size_t)n * R_REG + (c - C_IN)];   // action[b, r, :] contiguous since n = b*R + r
  } else {
    val = 0.f;
  }
  Acat[i] = f2bf(val);
}

// ---------------------------------------------------------------- pack weights to bf16 (W1 zero-padded to 192 rows)
__global__ void k_pack_w(const float* __restrict__ W1, const float* __restrict__ W2,
                         unsigned short* __restrict__ W1p, unsigned short* __restrict__ W2b) {
  int i = blockIdx.x * 256 + threadIdx.x;
  if (i < K1PAD * H_DIM) {
    int r = i / H_DIM;
    W1p[i] = (r < C_IN + R_REG) ? f2bf(W1[i]) : (unsigned short)0;
  } else if (i < K1PAD * H_DIM + H_DIM * H_DIM) {
    int j = i - K1PAD * H_DIM;
    W2b[j] = f2bf(W2[j]);
  }
}

// ---------------------------------------------------------------- GEMM1: [N,192] @ [192,256] -> relu -> h1 (bf16 WMMA)
__global__ __launch_bounds__(256) void k_gemm1(const unsigned* __restrict__ AcatU,
                                               const unsigned short* __restrict__ W1p,
                                               const float* __restrict__ b1,
                                               unsigned short* __restrict__ h1) {
  __shared__ unsigned Asl[128][20];  // 80B rows: 16B-aligned, conflict-free (20k mod 64 distinct)
  __shared__ unsigned Bsl[64][20];   // Bt[n][pair]
  int tid = threadIdx.x, lane = tid & 31, wave = tid >> 5;
  int rowBase = blockIdx.x * 128;
  int colBase = blockIdx.y * 64;
  v8f acc[4];
  #pragma unroll
  for (int t = 0; t < 4; ++t) acc[t] = (v8f){0, 0, 0, 0, 0, 0, 0, 0};

  for (int ks = 0; ks < 6; ++ks) {
    for (int i = tid; i < 128 * 4; i += 256) {       // A: 128 rows x 4x16B chunks, async to LDS
      int r = i >> 2, q = i & 3;
      async_copy16(AcatU + (size_t)(rowBase + r) * (K1PAD / 2) + ks * 16 + q * 4,
                   &Asl[r][q * 4]);
    }
    for (int i = tid; i < 64 * 16; i += 256) {       // B: pack bf16 pairs, transposed
      int n = i >> 4, p = i & 15;
      unsigned lo = W1p[(ks * 32 + 2 * p) * H_DIM + colBase + n];
      unsigned hi = W1p[(ks * 32 + 2 * p + 1) * H_DIM + colBase + n];
      Bsl[n][p] = lo | (hi << 16);
    }
    async_wait();
    __syncthreads();
    ABfrag a;
    int lm = wave * 16 + (lane & 15);
    int pa = (lane < 16) ? 0 : 4;
    #pragma unroll
    for (int i = 0; i < 8; ++i) a.u[i] = Asl[lm][pa + ((i < 4) ? i : i + 4)];
    int pb = (lane < 16) ? 0 : 4;
    #pragma unroll
    for (int nt = 0; nt < 4; ++nt) {
      ABfrag bf;
      int ln = nt * 16 + (lane & 15);
      #pragma unroll
      for (int i = 0; i < 8; ++i) bf.u[i] = Bsl[ln][pb + ((i < 4) ? i : i + 4)];
      acc[nt] = __builtin_amdgcn_wmma_f32_16x16x32_bf16(false, a.v, false, bf.v, (short)0, acc[nt], false, false);
    }
    __syncthreads();
  }
  #pragma unroll
  for (int nt = 0; nt < 4; ++nt) {
    int gn = colBase + nt * 16 + (lane & 15);
    float bias = b1[gn];
    #pragma unroll
    for (int r = 0; r < 8; ++r) {
      int gm = rowBase + wave * 16 + r + ((lane < 16) ? 0 : 8);
      h1[(size_t)gm * H_DIM + gn] = f2bf(fmaxf(acc[nt][r] + bias, 0.f));
    }
  }
}

// ---------------------------------------------------------------- GEMM2 + relu + masked region-pool + W3 dot (fused, per-graph)
__global__ __launch_bounds__(256) void k_gemm2_pool(const unsigned* __restrict__ h1U,
                                                    const unsigned short* __restrict__ W2b,
                                                    const float* __restrict__ b2,
                                                    const float* __restrict__ W3,
                                                    const float* __restrict__ b3,
                                                    float* __restrict__ out) {
  __shared__ unsigned Asl[112][20];   // 112 rows (100 valid, zero-pad) x K=32 chunk
  __shared__ unsigned Bsl[256][20];   // Bt[n][pair]
  __shared__ float pooled[256];
  __shared__ float red[256];
  int tid = threadIdx.x, lane = tid & 31, wave = tid >> 5;
  int b = blockIdx.x;
  size_t rowBase = (size_t)b * R_REG;
  pooled[tid] = 0.f;
  // zero the 12 padding rows once (never overwritten below)
  for (int i = tid; i < 12 * 16; i += 256) Asl[R_REG + (i >> 4)][i & 15] = 0u;

  v8f acc[14];
  #pragma unroll
  for (int t = 0; t < 14; ++t) acc[t] = (v8f){0, 0, 0, 0, 0, 0, 0, 0};

  for (int ks = 0; ks < 8; ++ks) {
    for (int i = tid; i < R_REG * 4; i += 256) {     // A: 100 valid rows x 4x16B, async to LDS
      int r = i >> 2, q = i & 3;
      async_copy16(h1U + (rowBase + r) * (H_DIM / 2) + ks * 16 + q * 4, &Asl[r][q * 4]);
    }
    for (int i = tid; i < 256 * 16; i += 256) {      // B: pack bf16 pairs, transposed
      int n = i >> 4, p = i & 15;
      unsigned lo = W2b[(ks * 32 + 2 * p) * H_DIM + n];
      unsigned hi = W2b[(ks * 32 + 2 * p + 1) * H_DIM + n];
      Bsl[n][p] = lo | (hi << 16);
    }
    async_wait();
    __syncthreads();
    ABfrag bf0, bf1;
    int pb = (lane < 16) ? 0 : 4;
    int n0 = (wave * 2 + 0) * 16 + (lane & 15);
    int n1 = (wave * 2 + 1) * 16 + (lane & 15);
    #pragma unroll
    for (int i = 0; i < 8; ++i) { bf0.u[i] = Bsl[n0][pb + ((i < 4) ? i : i + 4)]; bf1.u[i] = Bsl[n1][pb + ((i < 4) ? i : i + 4)]; }
    int pa = (lane < 16) ? 0 : 4;
    #pragma unroll
    for (int mt = 0; mt < 7; ++mt) {
      ABfrag a;
      int lm = mt * 16 + (lane & 15);
      #pragma unroll
      for (int i = 0; i < 8; ++i) a.u[i] = Asl[lm][pa + ((i < 4) ? i : i + 4)];
      acc[mt * 2 + 0] = __builtin_amdgcn_wmma_f32_16x16x32_bf16(false, a.v, false, bf0.v, (short)0, acc[mt * 2 + 0], false, false);
      acc[mt * 2 + 1] = __builtin_amdgcn_wmma_f32_16x16x32_bf16(false, a.v, false, bf1.v, (short)0, acc[mt * 2 + 1], false, false);
    }
    __syncthreads();
  }
  // bias + relu + pool (mask out padded rows m >= 100)
  #pragma unroll
  for (int t = 0; t < 2; ++t) {
    int gn = (wave * 2 + t) * 16 + (lane & 15);
    float bias = b2[gn];
    #pragma unroll
    for (int mt = 0; mt < 7; ++mt) {
      float part = 0.f;
      #pragma unroll
      for (int r = 0; r < 8; ++r) {
        int m = mt * 16 + r + ((lane < 16) ? 0 : 8);
        float v = fmaxf(acc[mt * 2 + t][r] + bias, 0.f);
        part += (m < R_REG) ? v : 0.f;
      }
      atomicAdd(&pooled[gn], part);
    }
  }
  __syncthreads();
  red[tid] = pooled[tid] * W3[tid];
  __syncthreads();
  for (int s = 128; s > 0; s >>= 1) {
    if (tid < s) red[tid] += red[tid + s];
    __syncthreads();
  }
  if (tid == 0) out[b] = red[0] + b3[0];
}

// ================================================================ launch
extern "C" void kernel_launch(void* const* d_in, const int* in_sizes, int n_in,
                              void* d_out, int out_size, void* d_ws, size_t ws_size,
                              hipStream_t stream) {
  const float*      state  = (const float*)d_in[0];
  const long long*  edge   = (const long long*)d_in[1];
  const float*      action = (const float*)d_in[2];
  const float*      W_gcn  = (const float*)d_in[3];
  const float*      b_gcn  = (const float*)d_in[4];
  const float*      W1     = (const float*)d_in[5];
  const float*      b1     = (const float*)d_in[6];
  const float*      W2     = (const float*)d_in[7];
  const float*      b2     = (const float*)d_in[8];
  const float*      W3     = (const float*)d_in[9];
  const float*      b3     = (const float*)d_in[10];
  float*            out    = (float*)d_out;
  (void)in_sizes; (void)n_in; (void)out_size; (void)ws_size;

  char* ws = (char*)d_ws;
  size_t off = 0;
  auto carve = [&](size_t bytes) { char* p = ws + off; off += (bytes + 255) & ~(size_t)255; return p; };
  float*          deg  = (float*)carve((size_t)N_NODES * 4);
  float*          xw   = (float*)carve((size_t)N_NODES * C_IN * 4);
  float*          agg  = (float*)carve((size_t)N_NODES * C_IN * 4);
  unsigned short* Acat = (unsigned short*)carve((size_t)N_NODES * K1PAD * 2);
  unsigned short* W1p  = (unsigned short*)carve((size_t)K1PAD * H_DIM * 2);
  unsigned short* W2b  = (unsigned short*)carve((size_t)H_DIM * H_DIM * 2);
  unsigned short* h1   = (unsigned short*)carve((size_t)N_NODES * H_DIM * 2);

  const long long* src = edge;
  const long long* dst = edge + E_EDGES;

  k_init      <<<(N_NODES * C_IN + 255) / 256, 256, 0, stream>>>(agg, deg);
  k_deg       <<<(E_EDGES + 255) / 256,        256, 0, stream>>>(dst, deg);
  k_dinv      <<<(N_NODES + 255) / 256,        256, 0, stream>>>(deg);
  k_xw        <<<N_NODES / 32,                 256, 0, stream>>>(state, W_gcn, xw);
  k_scatter   <<<(E_EDGES * 4 + 255) / 256,    256, 0, stream>>>(src, dst, deg, xw, agg);
  k_build_acat<<<(N_NODES * K1PAD + 255) / 256,256, 0, stream>>>(agg, xw, deg, b_gcn, state, action, Acat);
  k_pack_w    <<<(K1PAD * H_DIM + H_DIM * H_DIM + 255) / 256, 256, 0, stream>>>(W1, W2, W1p, W2b);
  k_gemm1     <<<dim3(N_NODES / 128, 4),       256, 0, stream>>>((const unsigned*)Acat, W1p, b1, h1);
  k_gemm2_pool<<<B_GRAPH,                      256, 0, stream>>>((const unsigned*)h1, W2b, b2, W3, b3, out);
}
